// LayerGIN_6957847020190
// MI455X (gfx1250) — matrix operand order repaired and checked by reading, not compiled, not run on hardware
//
#include <hip/hip_runtime.h>
#include <hip/hip_bf16.h>

#define D_IN  256
#define D_HID 512
#define D_OUT 256
#define LN_EPS 1e-5f

// Trivial ext-vector types (avoid HIP's class-based float4/uint4 in unions)
typedef __attribute__((ext_vector_type(16))) __bf16        v16bf;
typedef __attribute__((ext_vector_type(8)))  float         v8f;
typedef __attribute__((ext_vector_type(4)))  float         f32x4;
typedef __attribute__((ext_vector_type(4)))  unsigned int  u32x4;

union FragB16 {           // one WMMA 16-bit operand fragment: 16 bf16 per lane
    v16bf v;
    u32x4 q[2];
};

__device__ __forceinline__ unsigned int f32_to_bf16(float f) {
    unsigned int u = __float_as_uint(f);
    u += 0x7FFFu + ((u >> 16) & 1u);       // round-to-nearest-even
    return u >> 16;
}
__device__ __forceinline__ unsigned int pack_bf16x2(float lo, float hi) {
    return f32_to_bf16(lo) | (f32_to_bf16(hi) << 16);
}

// ---------------------------------------------------------------- weights -> bf16
__global__ void wconvert_kernel(const float* __restrict__ w1, const float* __restrict__ w2,
                                unsigned short* __restrict__ w1bf, unsigned short* __restrict__ w2bf) {
    int i = blockIdx.x * blockDim.x + threadIdx.x;
    int stride = gridDim.x * blockDim.x;
    for (int k = i; k < D_HID * D_IN; k += stride) w1bf[k] = (unsigned short)f32_to_bf16(w1[k]);
    for (int k = i; k < D_OUT * D_HID; k += stride) w2bf[k] = (unsigned short)f32_to_bf16(w2[k]);
}

// ---------------------------------------------------------------- agg = epsilon * v
__global__ void init_agg_kernel(const f32x4* __restrict__ v, const float* __restrict__ eps,
                                f32x4* __restrict__ agg, int n4) {
    float e = eps[0];
    int i = blockIdx.x * blockDim.x + threadIdx.x;
    int stride = gridDim.x * blockDim.x;
    for (int k = i; k < n4; k += stride) agg[k] = v[k] * e;
}

// ---------------------------------------------------------------- agg[dst] += w * v[src]
// One wave32 per edge; lane handles 8 strided columns (coalesced 128B bursts).
// v (51MB) and agg (51MB) both L2-resident on MI455X's 192MB L2.
__global__ void scatter_kernel(const float* __restrict__ v, const float* __restrict__ ew,
                               const int* __restrict__ src, const int* __restrict__ dst,
                               float* __restrict__ agg, int E) {
    int gid = blockIdx.x * blockDim.x + threadIdx.x;
    int e = gid >> 5;
    int lane = gid & 31;
    if (e >= E) return;
    int s = src[e], d = dst[e];
    float w = ew[e];
    const float* vs = v + (size_t)s * D_IN;
    float* ad = agg + (size_t)d * D_IN;
#pragma unroll
    for (int j = 0; j < D_IN / 32; ++j) {
        int c = lane + j * 32;
        __hip_atomic_fetch_add(&ad[c], w * vs[c], __ATOMIC_RELAXED, __HIP_MEMORY_SCOPE_AGENT);
    }
}

// ---------------------------------------------------------------- GEMM1 + bias + LN + ReLU
// Block: 256 threads (8 waves), tile = 16 rows x 512 cols, K = 256 (8 WMMA steps).
// Each wave owns 4 16x16 output tiles.
#define A1_STRIDE 264   // 256 + 8 ushorts  (bank-conflict padding)
#define C1_STRIDE 516   // 512 + 4 floats

__global__ void __launch_bounds__(256)
gemm1_ln_relu_kernel(const float* __restrict__ agg, const unsigned short* __restrict__ w1bf,
                     const float* __restrict__ b1, const float* __restrict__ g1,
                     const float* __restrict__ beta1, unsigned short* __restrict__ h1) {
    __shared__ __align__(16) unsigned short Abuf[16 * A1_STRIDE];
    __shared__ float Cbuf[16 * C1_STRIDE];

    const int tid  = threadIdx.x;
    const int lane = tid & 31;
    const int wv   = tid >> 5;
    const int row0 = blockIdx.x * 16;

    // Stage A tile: 16 rows x 256 f32 -> bf16 in LDS (each thread: 16 contiguous cols)
    {
        const int r  = tid >> 4;
        const int c0 = (tid & 15) * 16;
        const f32x4* sp = (const f32x4*)(agg + (size_t)(row0 + r) * D_IN + c0);
        f32x4 qa = sp[0], qb = sp[1], qc = sp[2], qd = sp[3];
        u32x4 lo, hi;
        lo[0] = pack_bf16x2(qa[0], qa[1]); lo[1] = pack_bf16x2(qa[2], qa[3]);
        lo[2] = pack_bf16x2(qb[0], qb[1]); lo[3] = pack_bf16x2(qb[2], qb[3]);
        hi[0] = pack_bf16x2(qc[0], qc[1]); hi[1] = pack_bf16x2(qc[2], qc[3]);
        hi[2] = pack_bf16x2(qd[0], qd[1]); hi[3] = pack_bf16x2(qd[2], qd[3]);
        u32x4* dp = (u32x4*)&Abuf[r * A1_STRIDE + c0];
        dp[0] = lo; dp[1] = hi;
    }
    __syncthreads();

    v8f acc[4];
    const v8f vzero = {0.f, 0.f, 0.f, 0.f, 0.f, 0.f, 0.f, 0.f};
#pragma unroll
    for (int t = 0; t < 4; ++t) acc[t] = vzero;

    const int rsel  = lane & 15;
    const int khalf = (lane >> 4) * 16;    // A: 16 consecutive K per lane half
    const int kseg  = (lane >> 4) * 16;    // B: 16 consecutive K per lane half
    const int nw    = wv * 64;

#pragma unroll
    for (int ks = 0; ks < 8; ++ks) {
        const int kb = ks * 32;
        FragB16 a;
        a.q[0] = *(const u32x4*)&Abuf[rsel * A1_STRIDE + kb + khalf];
        a.q[1] = *(const u32x4*)&Abuf[rsel * A1_STRIDE + kb + 8 + khalf];
#pragma unroll
        for (int t = 0; t < 4; ++t) {
            const int ncol = nw + t * 16 + rsel;          // B col = w1 row (out feature)
            const u32x4* bp = (const u32x4*)&w1bf[ncol * D_IN + kb + kseg];
            FragB16 b;
            b.q[0] = bp[0]; b.q[1] = bp[1];
            acc[t] = __builtin_amdgcn_wmma_f32_16x16x32_bf16(
                false, a.v, false, b.v, (short)0, acc[t], false, false);
        }
    }

    // D layout: VGPR i -> row (lane>=16 ? 8 : 0)+i, col = lane&15
    const int mbase = (lane >> 4) * 8;
#pragma unroll
    for (int t = 0; t < 4; ++t) {
        const int c = nw + t * 16 + rsel;
#pragma unroll
        for (int i = 0; i < 8; ++i)
            Cbuf[(mbase + i) * C1_STRIDE + c] = acc[t][i];
    }
    __syncthreads();

    // LayerNorm + ReLU: wave handles 2 rows, 16 elems per lane, wave32 shfl reduce.
#pragma unroll
    for (int rr = 0; rr < 2; ++rr) {
        const int r = wv * 2 + rr;
        float x[16], s1 = 0.f, s2 = 0.f;
#pragma unroll
        for (int j = 0; j < 16; ++j) {
            const int c = lane + j * 32;
            float t = Cbuf[r * C1_STRIDE + c] + b1[c];
            x[j] = t; s1 += t; s2 += t * t;
        }
#pragma unroll
        for (int off = 16; off >= 1; off >>= 1) {
            s1 += __shfl_xor(s1, off, 32);
            s2 += __shfl_xor(s2, off, 32);
        }
        const float mu  = s1 * (1.0f / D_HID);
        const float var = s2 * (1.0f / D_HID) - mu * mu;
        const float rs  = rsqrtf(var + LN_EPS);
        unsigned short* op = h1 + (size_t)(row0 + r) * D_HID;
#pragma unroll
        for (int j = 0; j < 16; ++j) {
            const int c = lane + j * 32;
            float y = (x[j] - mu) * rs * g1[c] + beta1[c];
            op[c] = (unsigned short)f32_to_bf16(fmaxf(y, 0.0f));
        }
    }
}

// ---------------------------------------------------------------- GEMM2 + bias + LN + ReLU
// Block: 256 threads, tile = 16 rows x 256 cols, K = 512 (16 WMMA steps); 2 tiles/wave.
#define A2_STRIDE 520   // 512 + 8 ushorts
#define C2_STRIDE 260   // 256 + 4 floats

__global__ void __launch_bounds__(256)
gemm2_ln_relu_kernel(const unsigned short* __restrict__ h1, const unsigned short* __restrict__ w2bf,
                     const float* __restrict__ b2, const float* __restrict__ g2,
                     const float* __restrict__ beta2, float* __restrict__ out) {
    __shared__ __align__(16) unsigned short Abuf[16 * A2_STRIDE];
    __shared__ float Cbuf[16 * C2_STRIDE];

    const int tid  = threadIdx.x;
    const int lane = tid & 31;
    const int wv   = tid >> 5;
    const int row0 = blockIdx.x * 16;

    // Stage A tile: 16 rows x 512 bf16 (each thread: 32 contiguous cols = 64B)
    {
        const int r  = tid >> 4;
        const int c0 = (tid & 15) * 32;
        const u32x4* sp = (const u32x4*)(h1 + (size_t)(row0 + r) * D_HID + c0);
        u32x4* dp = (u32x4*)&Abuf[r * A2_STRIDE + c0];
        dp[0] = sp[0]; dp[1] = sp[1]; dp[2] = sp[2]; dp[3] = sp[3];
    }
    __syncthreads();

    const v8f vzero = {0.f, 0.f, 0.f, 0.f, 0.f, 0.f, 0.f, 0.f};
    v8f acc[2] = {vzero, vzero};
    const int rsel  = lane & 15;
    const int khalf = (lane >> 4) * 16;
    const int kseg  = (lane >> 4) * 16;
    const int nw    = wv * 32;

#pragma unroll
    for (int ks = 0; ks < 16; ++ks) {
        const int kb = ks * 32;
        FragB16 a;
        a.q[0] = *(const u32x4*)&Abuf[rsel * A2_STRIDE + kb + khalf];
        a.q[1] = *(const u32x4*)&Abuf[rsel * A2_STRIDE + kb + 8 + khalf];
#pragma unroll
        for (int t = 0; t < 2; ++t) {
            const int ncol = nw + t * 16 + rsel;
            const u32x4* bp = (const u32x4*)&w2bf[ncol * D_HID + kb + kseg];
            FragB16 b;
            b.q[0] = bp[0]; b.q[1] = bp[1];
            acc[t] = __builtin_amdgcn_wmma_f32_16x16x32_bf16(
                false, a.v, false, b.v, (short)0, acc[t], false, false);
        }
    }

    const int mbase = (lane >> 4) * 8;
#pragma unroll
    for (int t = 0; t < 2; ++t) {
        const int c = nw + t * 16 + rsel;
#pragma unroll
        for (int i = 0; i < 8; ++i)
            Cbuf[(mbase + i) * C2_STRIDE + c] = acc[t][i];
    }
    __syncthreads();

#pragma unroll
    for (int rr = 0; rr < 2; ++rr) {
        const int r = wv * 2 + rr;
        float x[8], s1 = 0.f, s2 = 0.f;
#pragma unroll
        for (int j = 0; j < 8; ++j) {
            const int c = lane + j * 32;
            float t = Cbuf[r * C2_STRIDE + c] + b2[c];
            x[j] = t; s1 += t; s2 += t * t;
        }
#pragma unroll
        for (int off = 16; off >= 1; off >>= 1) {
            s1 += __shfl_xor(s1, off, 32);
            s2 += __shfl_xor(s2, off, 32);
        }
        const float mu  = s1 * (1.0f / D_OUT);
        const float var = s2 * (1.0f / D_OUT) - mu * mu;
        const float rs  = rsqrtf(var + LN_EPS);
        float* op = out + (size_t)(row0 + r) * D_OUT;
#pragma unroll
        for (int j = 0; j < 8; ++j) {
            const int c = lane + j * 32;
            float y = (x[j] - mu) * rs * g2[c] + beta2[c];
            op[c] = fmaxf(y, 0.0f);
        }
    }
}

// ----------------------------------------------------------------
extern "C" void kernel_launch(void* const* d_in, const int* in_sizes, int n_in,
                              void* d_out, int out_size, void* d_ws, size_t ws_size,
                              hipStream_t stream) {
    const float* v     = (const float*)d_in[0];
    const float* ew    = (const float*)d_in[1];
    const float* eps   = (const float*)d_in[2];
    const float* w1    = (const float*)d_in[3];
    const float* b1    = (const float*)d_in[4];
    const float* g1    = (const float*)d_in[5];
    const float* beta1 = (const float*)d_in[6];
    const float* w2    = (const float*)d_in[7];
    const float* b2    = (const float*)d_in[8];
    const float* g2    = (const float*)d_in[9];
    const float* beta2 = (const float*)d_in[10];
    const int*   src   = (const int*)d_in[11];
    const int*   dst   = (const int*)d_in[12];
    float*       out   = (float*)d_out;

    const int N = in_sizes[0] / D_IN;   // 50000 (multiple of 16)
    const int E = in_sizes[1];          // 800000

    // Workspace layout (all fully rewritten each call -> deterministic):
    //   agg  f32  [N][256]   51.2 MB
    //   h1   bf16 [N][512]   51.2 MB
    //   w1bf bf16 [512][256] 256 KB
    //   w2bf bf16 [256][512] 256 KB
    char* ws = (char*)d_ws;
    float*          agg  = (float*)ws;
    unsigned short* h1   = (unsigned short*)(ws + (size_t)N * D_IN * sizeof(float));
    unsigned short* w1bf = (unsigned short*)(ws + (size_t)N * D_IN * sizeof(float)
                                                + (size_t)N * D_HID * sizeof(unsigned short));
    unsigned short* w2bf = w1bf + D_HID * D_IN;

    wconvert_kernel<<<256, 256, 0, stream>>>(w1, w2, w1bf, w2bf);

    const int n4 = N * D_IN / 4;
    init_agg_kernel<<<2048, 256, 0, stream>>>((const f32x4*)v, eps, (f32x4*)agg, n4);

    const int sblocks = (int)(((long long)E * 32 + 255) / 256);
    scatter_kernel<<<sblocks, 256, 0, stream>>>(v, ew, src, dst, agg, E);

    gemm1_ln_relu_kernel<<<N / 16, 256, 0, stream>>>(agg, w1bf, b1, g1, beta1, h1);
    gemm2_ln_relu_kernel<<<N / 16, 256, 0, stream>>>(h1, w2bf, b2, g2, beta2, out);
}
